// MAGNET_9036611190793
// MI455X (gfx1250) — compile-verified, matrix operand order checked by reading, not run on hardware
//
#include <hip/hip_runtime.h>
#include <hip/hip_bf16.h>
#include <math.h>

typedef __attribute__((ext_vector_type(16))) __bf16 v16bf;
typedef __attribute__((ext_vector_type(8)))  float  v8f;

#define BB    2
#define NQC   4
#define TT    1024
#define DDIM  1024
#define HH    16
#define HDIM  64
#define LLAY  8
#define CARDN 1024
#define NCOND 64
#define CONDD 768
#define WINW  5
#define DFFN  4096
#define BT    (BB*TT)

// gfx1250 async global->LDS path (ASYNCcnt-tracked, bypasses VGPRs).
// Device-pass only so the host pass never sees the address-space casts.
#if defined(__gfx1250__) && __has_builtin(__builtin_amdgcn_global_load_async_to_lds_b128)
#define USE_ASYNC_LDS 1
#else
#define USE_ASYNC_LDS 0
#endif

#if USE_ASYNC_LDS
typedef int v4i_as __attribute__((vector_size(16)));
typedef __attribute__((address_space(1))) v4i_as* gv4p;
typedef __attribute__((address_space(3))) v4i_as* lv4p;
__device__ inline void async_cp16(const __bf16* g, __bf16* l) {
  // GLOBAL_LOAD_ASYNC_TO_LDS_B128: LDS[l] = MEM[g], 16B per lane
  __builtin_amdgcn_global_load_async_to_lds_b128((gv4p)g, (lv4p)l, 0, 0);
}
__device__ inline void wait_async0() {
#if __has_builtin(__builtin_amdgcn_s_wait_asynccnt)
  __builtin_amdgcn_s_wait_asynccnt(0);
#else
  asm volatile("s_wait_asynccnt 0x0" ::: "memory");
#endif
}
#endif

struct Frag32 { uint4 lo, hi; };

__device__ inline float wredsum(float x) {
#pragma unroll
  for (int o = 16; o > 0; o >>= 1) x += __shfl_xor(x, o, 32);
  return x;
}

// ---------------------------------------------------------------------------
// Tiled bf16 WMMA GEMM:  C(MxN) = A(MxK) * Wt^T + bias
//   A  : M x K row-major bf16
//   Wt : N x K row-major bf16 (weights pre-transposed)
// 128x128 block tile, 8 waves (each 64x32 = 4x2 WMMA tiles), K-step 32,
// double-buffered LDS; tile k+1 streams in (async-to-LDS on gfx1250) while
// WMMA consumes tile k. One barrier per K-step.
// Epilogue: RES -> Cf += v ; WF32 -> Cf = v ; WB16 -> Cb = bf16(v); opt GELU.
// ---------------------------------------------------------------------------
template <bool RES, bool GELU, bool WF32, bool WB16>
__global__ __launch_bounds__(256) void gemm_bf16_kernel(
    const __bf16* __restrict__ A, const __bf16* __restrict__ Wt,
    const float* __restrict__ bias, float* __restrict__ Cf,
    __bf16* __restrict__ Cb, int M, int N, int K) {
  __shared__ __bf16 sA[2][128][40];  // rows padded to 40 (80B = 5x16B)
  __shared__ __bf16 sB[2][128][40];
  const int tid = threadIdx.x;
  const int lane = tid & 31, wave = tid >> 5;
  const int waveM = wave >> 2, waveN = wave & 3;
  const int bm = blockIdx.y * 128, bn = blockIdx.x * 128;
  const int sel = lane >> 4, nl = lane & 15;

  v8f acc[4][2];
#pragma unroll
  for (int i = 0; i < 4; i++)
#pragma unroll
    for (int j = 0; j < 2; j++) acc[i][j] = 0.0f;

  const int srow = tid >> 1, shalf = tid & 1;
  const __bf16* Aptr = A + (size_t)(bm + srow) * K + shalf * 16;
  const __bf16* Bptr = Wt + (size_t)(bn + srow) * K + shalf * 16;

  // ---- prologue: stage K-tile 0 into buffer 0 ----
#if USE_ASYNC_LDS
  async_cp16(Aptr,     &sA[0][srow][shalf * 16]);
  async_cp16(Aptr + 8, &sA[0][srow][shalf * 16 + 8]);
  async_cp16(Bptr,     &sB[0][srow][shalf * 16]);
  async_cp16(Bptr + 8, &sB[0][srow][shalf * 16 + 8]);
  wait_async0();
  __syncthreads();
#else
  {
    uint4 ra0 = *(const uint4*)(Aptr);
    uint4 ra1 = *(const uint4*)(Aptr + 8);
    uint4 rb0 = *(const uint4*)(Bptr);
    uint4 rb1 = *(const uint4*)(Bptr + 8);
    *(uint4*)&sA[0][srow][shalf * 16] = ra0;
    *(uint4*)&sA[0][srow][shalf * 16 + 8] = ra1;
    *(uint4*)&sB[0][srow][shalf * 16] = rb0;
    *(uint4*)&sB[0][srow][shalf * 16 + 8] = rb1;
    __syncthreads();
  }
#endif

  int buf = 0;
  for (int k0 = 0; k0 < K; k0 += 32) {
    const bool more = (k0 + 32) < K;
#if USE_ASYNC_LDS
    if (more) {  // stream next tile into the other buffer while we compute
      async_cp16(Aptr + k0 + 32, &sA[buf ^ 1][srow][shalf * 16]);
      async_cp16(Aptr + k0 + 40, &sA[buf ^ 1][srow][shalf * 16 + 8]);
      async_cp16(Bptr + k0 + 32, &sB[buf ^ 1][srow][shalf * 16]);
      async_cp16(Bptr + k0 + 40, &sB[buf ^ 1][srow][shalf * 16 + 8]);
    }
#else
    uint4 na0 = {}, na1 = {}, nb0 = {}, nb1 = {};
    if (more) {
      __builtin_prefetch(Aptr + k0 + 64, 0, 3);
      __builtin_prefetch(Bptr + k0 + 64, 0, 3);
      na0 = *(const uint4*)(Aptr + k0 + 32);
      na1 = *(const uint4*)(Aptr + k0 + 40);
      nb0 = *(const uint4*)(Bptr + k0 + 32);
      nb1 = *(const uint4*)(Bptr + k0 + 40);
    }
#endif

    // fragment loads (ISA 16-bit A-matrix layout: lanes 0-15 K0-7/16-23,
    // lanes 16-31 K8-15/24-31 -> two contiguous 16B chunks per lane)
    v16bf af[4], bfr[2];
#pragma unroll
    for (int mt = 0; mt < 4; mt++) {
      int r = waveM * 64 + mt * 16 + nl;
      Frag32 f;
      f.lo = *(const uint4*)&sA[buf][r][sel * 8];
      f.hi = *(const uint4*)&sA[buf][r][16 + sel * 8];
      af[mt] = __builtin_bit_cast(v16bf, f);
    }
#pragma unroll
    for (int nt = 0; nt < 2; nt++) {
      int r = waveN * 32 + nt * 16 + nl;
      Frag32 f;
      f.lo = *(const uint4*)&sB[buf][r][sel * 8];
      f.hi = *(const uint4*)&sB[buf][r][16 + sel * 8];
      bfr[nt] = __builtin_bit_cast(v16bf, f);
    }
#pragma unroll
    for (int mt = 0; mt < 4; mt++)
#pragma unroll
      for (int nt = 0; nt < 2; nt++)
        acc[mt][nt] = __builtin_amdgcn_wmma_f32_16x16x32_bf16(
            false, af[mt], false, bfr[nt], (short)0, acc[mt][nt], false, false);

#if USE_ASYNC_LDS
    if (more) {
      wait_async0();     // our async writes to buf^1 have landed
      __syncthreads();   // everyone's writes visible, all reads of buf done
    }
#else
    if (more) {
      *(uint4*)&sA[buf ^ 1][srow][shalf * 16] = na0;
      *(uint4*)&sA[buf ^ 1][srow][shalf * 16 + 8] = na1;
      *(uint4*)&sB[buf ^ 1][srow][shalf * 16] = nb0;
      *(uint4*)&sB[buf ^ 1][srow][shalf * 16 + 8] = nb1;
      __syncthreads();
    }
#endif
    buf ^= 1;
  }

#pragma unroll
  for (int mt = 0; mt < 4; mt++) {
    int row = bm + waveM * 64 + mt * 16 + sel * 8;
#pragma unroll
    for (int nt = 0; nt < 2; nt++) {
      int col = bn + waveN * 32 + nt * 16 + nl;
      float bv = bias ? bias[col] : 0.0f;
#pragma unroll
      for (int r = 0; r < 8; r++) {
        float v = acc[mt][nt][r] + bv;
        if (GELU) {
          float u = 0.7978845608028654f * (v + 0.044715f * v * v * v);
          v = 0.5f * v * (1.0f + tanhf(u));
        }
        size_t idx = (size_t)(row + r) * N + col;
        if (RES)
          Cf[idx] += v;
        else if (WF32)
          Cf[idx] = v;
        if (WB16) Cb[idx] = (__bf16)v;
      }
    }
  }
}

// ---------------------------------------------------------------------------
// LayerNorm: fp32 in -> bf16 out, one 256-thread block per row (D = 1024)
// ---------------------------------------------------------------------------
__global__ __launch_bounds__(256) void ln_kernel(const float* __restrict__ x,
                                                 const float* __restrict__ g,
                                                 const float* __restrict__ b,
                                                 __bf16* __restrict__ y) {
  const int row = blockIdx.x;
  const float* xr = x + (size_t)row * DDIM;
  float r[4], s = 0.f, s2 = 0.f;
#pragma unroll
  for (int j = 0; j < 4; j++) {
    float v = xr[threadIdx.x + j * 256];
    r[j] = v;
    s += v;
    s2 += v * v;
  }
  s = wredsum(s);
  s2 = wredsum(s2);
  __shared__ float red0[8], red1[8];
  if ((threadIdx.x & 31) == 0) {
    red0[threadIdx.x >> 5] = s;
    red1[threadIdx.x >> 5] = s2;
  }
  __syncthreads();
  float t0 = 0.f, t1 = 0.f;
#pragma unroll
  for (int w = 0; w < 8; w++) {
    t0 += red0[w];
    t1 += red1[w];
  }
  float mean = t0 * (1.0f / DDIM);
  float var = t1 * (1.0f / DDIM) - mean * mean;
  float rs = rsqrtf(var + 1e-5f);
#pragma unroll
  for (int j = 0; j < 4; j++) {
    int d = threadIdx.x + j * 256;
    y[(size_t)row * DDIM + d] = (__bf16)((r[j] - mean) * rs * g[d] + b[d]);
  }
}

// ---------------------------------------------------------------------------
// Embedding sum + interleaved sin/cos positional embedding
// ---------------------------------------------------------------------------
__global__ __launch_bounds__(256) void embed_kernel(const int* __restrict__ x,
                                                    const float* __restrict__ emb,
                                                    float* __restrict__ h) {
  const int row = blockIdx.x;  // b*T + t
  const int b = row / TT, t = row % TT;
  int idx[NQC];
#pragma unroll
  for (int q = 0; q < NQC; q++) idx[q] = x[(b * NQC + q) * TT + t];
  const float ln10k = 9.210340371976184f;  // log(10000)
  for (int d = threadIdx.x; d < DDIM; d += 256) {
    float a = 0.f;
#pragma unroll
    for (int q = 0; q < NQC; q++)
      a += emb[((size_t)q * (CARDN + 1) + idx[q]) * DDIM + d];
    int i2 = d >> 1;
    float theta = (float)t * __expf(-(float)(2 * i2) * (ln10k / DDIM));
    a += (d & 1) ? __cosf(theta) : __sinf(theta);
    h[(size_t)row * DDIM + d] = a;
  }
}

__global__ void f32_to_bf16_kernel(const float* __restrict__ in,
                                   __bf16* __restrict__ out, int n) {
  for (int i = blockIdx.x * blockDim.x + threadIdx.x; i < n;
       i += gridDim.x * blockDim.x)
    out[i] = (__bf16)in[i];
}

// Transpose + convert: W (KxN fp32, row-major) -> Wt (NxK bf16, row-major)
__global__ __launch_bounds__(256) void tconv_kernel(const float* __restrict__ W,
                                                    __bf16* __restrict__ Wt,
                                                    int K, int N) {
  __shared__ float tile[32][33];
  int n0 = blockIdx.x * 32, k0 = blockIdx.y * 32;
#pragma unroll
  for (int j = 0; j < 4; j++) {
    int r = threadIdx.y + j * 8;
    tile[r][threadIdx.x] = W[(size_t)(k0 + r) * N + n0 + threadIdx.x];
  }
  __syncthreads();
#pragma unroll
  for (int j = 0; j < 4; j++) {
    int r = threadIdx.y + j * 8;
    Wt[(size_t)(n0 + r) * K + k0 + threadIdx.x] = (__bf16)tile[threadIdx.x][r];
  }
}

// ---------------------------------------------------------------------------
// Banded self-attention (window +-5): one wave32 per (b, h, t); 2 dims/lane
// ---------------------------------------------------------------------------
__global__ __launch_bounds__(256) void self_attn_kernel(
    const float* __restrict__ q, const float* __restrict__ k,
    const float* __restrict__ v, float* __restrict__ o) {
  const int lane = threadIdx.x & 31;
  const int gid = blockIdx.x * 8 + (threadIdx.x >> 5);
  const int t = gid % TT, hh = (gid / TT) % HH, b = gid / (TT * HH);
  const int dd = lane * 2;
  const size_t qoff = ((size_t)(b * TT + t)) * DDIM + hh * HDIM + dd;
  float2 qv = *(const float2*)(q + qoff);
  float sc[2 * WINW + 1];
#pragma unroll
  for (int j = 0; j < 2 * WINW + 1; j++) {
    int s = t - WINW + j;
    float val = -__builtin_inff();
    if (s >= 0 && s < TT) {
      float2 kv = *(const float2*)(k + ((size_t)(b * TT + s)) * DDIM + hh * HDIM + dd);
      float p = qv.x * kv.x + qv.y * kv.y;
      val = wredsum(p) * 0.125f;  // 1/sqrt(64)
    }
    sc[j] = val;
  }
  float mx = sc[0];
#pragma unroll
  for (int j = 1; j < 2 * WINW + 1; j++) mx = fmaxf(mx, sc[j]);
  float ss = 0.f;
#pragma unroll
  for (int j = 0; j < 2 * WINW + 1; j++) {
    float e = __expf(sc[j] - mx);
    sc[j] = e;
    ss += e;
  }
  float ax = 0.f, ay = 0.f;
#pragma unroll
  for (int j = 0; j < 2 * WINW + 1; j++) {
    int s = t - WINW + j;
    if (s >= 0 && s < TT) {
      float2 vv = *(const float2*)(v + ((size_t)(b * TT + s)) * DDIM + hh * HDIM + dd);
      ax += sc[j] * vv.x;
      ay += sc[j] * vv.y;
    }
  }
  float inv = 1.0f / ss;
  float2 res;
  res.x = ax * inv;
  res.y = ay * inv;
  *(float2*)(o + qoff) = res;
}

// ---------------------------------------------------------------------------
// Cross-attention over 64 cond positions: one wave32 per (b, h, t)
// ---------------------------------------------------------------------------
__global__ __launch_bounds__(256) void cross_attn_kernel(
    const float* __restrict__ q, const float* __restrict__ kc,
    const float* __restrict__ vc, float* __restrict__ o) {
  const int lane = threadIdx.x & 31;
  const int gid = blockIdx.x * 8 + (threadIdx.x >> 5);
  const int t = gid % TT, hh = (gid / TT) % HH, b = gid / (TT * HH);
  const int dd = lane * 2;
  const size_t qoff = ((size_t)(b * TT + t)) * DDIM + hh * HDIM + dd;
  float2 qv = *(const float2*)(q + qoff);
  float sc[NCOND];
#pragma unroll
  for (int j = 0; j < NCOND; j++) {
    float2 kv = *(const float2*)(kc + ((size_t)(b * NCOND + j)) * DDIM + hh * HDIM + dd);
    float p = qv.x * kv.x + qv.y * kv.y;
    sc[j] = wredsum(p) * 0.125f;
  }
  float mx = sc[0];
#pragma unroll
  for (int j = 1; j < NCOND; j++) mx = fmaxf(mx, sc[j]);
  float ss = 0.f;
#pragma unroll
  for (int j = 0; j < NCOND; j++) {
    float e = __expf(sc[j] - mx);
    sc[j] = e;
    ss += e;
  }
  float ax = 0.f, ay = 0.f;
#pragma unroll
  for (int j = 0; j < NCOND; j++) {
    float2 vv = *(const float2*)(vc + ((size_t)(b * NCOND + j)) * DDIM + hh * HDIM + dd);
    ax += sc[j] * vv.x;
    ay += sc[j] * vv.y;
  }
  float inv = 1.0f / ss;
  float2 res;
  res.x = ax * inv;
  res.y = ay * inv;
  *(float2*)(o + qoff) = res;
}

// ---------------------------------------------------------------------------
extern "C" void kernel_launch(void* const* d_in, const int* in_sizes, int n_in,
                              void* d_out, int out_size, void* d_ws,
                              size_t ws_size, hipStream_t stream) {
  (void)in_sizes; (void)n_in; (void)out_size; (void)ws_size;
  const int*   x     = (const int*)d_in[0];
  const float* cond  = (const float*)d_in[1];
  const float* emb   = (const float*)d_in[2];
  const float* ln1_g = (const float*)d_in[3];
  const float* ln1_b = (const float*)d_in[4];
  const float* ln2_g = (const float*)d_in[5];
  const float* ln2_b = (const float*)d_in[6];
  const float* ln3_g = (const float*)d_in[7];
  const float* ln3_b = (const float*)d_in[8];
  const float* sWq = (const float*)d_in[9];
  const float* sWk = (const float*)d_in[10];
  const float* sWv = (const float*)d_in[11];
  const float* sWo = (const float*)d_in[12];
  const float* sbq = (const float*)d_in[13];
  const float* sbk = (const float*)d_in[14];
  const float* sbv = (const float*)d_in[15];
  const float* sbo = (const float*)d_in[16];
  const float* cWq = (const float*)d_in[17];
  const float* cWk = (const float*)d_in[18];
  const float* cWv = (const float*)d_in[19];
  const float* cWo = (const float*)d_in[20];
  const float* cbq = (const float*)d_in[21];
  const float* cbk = (const float*)d_in[22];
  const float* cbv = (const float*)d_in[23];
  const float* cbo = (const float*)d_in[24];
  const float* W1  = (const float*)d_in[25];
  const float* b1  = (const float*)d_in[26];
  const float* W2  = (const float*)d_in[27];
  const float* b2  = (const float*)d_in[28];
  const float* outg = (const float*)d_in[29];
  const float* outb = (const float*)d_in[30];
  const float* headW = (const float*)d_in[31];
  const float* headb = (const float*)d_in[32];
  float* out = (float*)d_out;

  char* base = (char*)d_ws;
  size_t off = 0;
  auto alloc = [&](size_t bytes) -> char* {
    char* p = base + off;
    off += (bytes + 255) & ~(size_t)255;
    return p;
  };
  float*  h     = (float*)alloc((size_t)BT * DDIM * 4);
  __bf16* lnb   = (__bf16*)alloc((size_t)BT * DDIM * 2);
  float*  qf    = (float*)alloc((size_t)BT * DDIM * 4);
  float*  kf    = (float*)alloc((size_t)BT * DDIM * 4);
  float*  vf    = (float*)alloc((size_t)BT * DDIM * 4);
  float*  kcf   = (float*)alloc((size_t)BB * NCOND * DDIM * 4);
  float*  vcf   = (float*)alloc((size_t)BB * NCOND * DDIM * 4);
  float*  attf  = (float*)alloc((size_t)BT * DDIM * 4);
  __bf16* attb  = (__bf16*)alloc((size_t)BT * DDIM * 2);
  __bf16* ffnb  = (__bf16*)alloc((size_t)BT * DFFN * 2);
  __bf16* condb = (__bf16*)alloc((size_t)BB * NCOND * CONDD * 2);
  __bf16* wt    = (__bf16*)alloc((size_t)DDIM * DFFN * 2);  // staged bf16 N-major weights

  auto tconv = [&](const float* W, int K, int N) {
    tconv_kernel<<<dim3(N / 32, K / 32), dim3(32, 8), 0, stream>>>(W, wt, K, N);
  };

  embed_kernel<<<BT, 256, 0, stream>>>(x, emb, h);
  f32_to_bf16_kernel<<<256, 256, 0, stream>>>(cond, condb, BB * NCOND * CONDD);

  const int AT = (BB * HH * TT) / 8;  // attention waves / 8 per block

  for (int l = 0; l < LLAY; l++) {
    // ---- banded self attention ----
    ln_kernel<<<BT, 256, 0, stream>>>(h, ln1_g + l * DDIM, ln1_b + l * DDIM, lnb);
    tconv(sWq + (size_t)l * DDIM * DDIM, DDIM, DDIM);
    gemm_bf16_kernel<false, false, true, false><<<dim3(DDIM / 128, BT / 128), 256, 0, stream>>>(
        lnb, wt, sbq + l * DDIM, qf, nullptr, BT, DDIM, DDIM);
    tconv(sWk + (size_t)l * DDIM * DDIM, DDIM, DDIM);
    gemm_bf16_kernel<false, false, true, false><<<dim3(DDIM / 128, BT / 128), 256, 0, stream>>>(
        lnb, wt, sbk + l * DDIM, kf, nullptr, BT, DDIM, DDIM);
    tconv(sWv + (size_t)l * DDIM * DDIM, DDIM, DDIM);
    gemm_bf16_kernel<false, false, true, false><<<dim3(DDIM / 128, BT / 128), 256, 0, stream>>>(
        lnb, wt, sbv + l * DDIM, vf, nullptr, BT, DDIM, DDIM);
    self_attn_kernel<<<AT, 256, 0, stream>>>(qf, kf, vf, attf);
    f32_to_bf16_kernel<<<1024, 256, 0, stream>>>(attf, attb, BT * DDIM);
    tconv(sWo + (size_t)l * DDIM * DDIM, DDIM, DDIM);
    gemm_bf16_kernel<true, false, true, false><<<dim3(DDIM / 128, BT / 128), 256, 0, stream>>>(
        attb, wt, sbo + l * DDIM, h, nullptr, BT, DDIM, DDIM);

    // ---- cross attention (64 cond keys) ----
    ln_kernel<<<BT, 256, 0, stream>>>(h, ln2_g + l * DDIM, ln2_b + l * DDIM, lnb);
    tconv(cWq + (size_t)l * DDIM * DDIM, DDIM, DDIM);
    gemm_bf16_kernel<false, false, true, false><<<dim3(DDIM / 128, BT / 128), 256, 0, stream>>>(
        lnb, wt, cbq + l * DDIM, qf, nullptr, BT, DDIM, DDIM);
    tconv(cWk + (size_t)l * CONDD * DDIM, CONDD, DDIM);
    gemm_bf16_kernel<false, false, true, false><<<dim3(DDIM / 128, (BB * NCOND) / 128), 256, 0, stream>>>(
        condb, wt, cbk + l * DDIM, kcf, nullptr, BB * NCOND, DDIM, CONDD);
    tconv(cWv + (size_t)l * CONDD * DDIM, CONDD, DDIM);
    gemm_bf16_kernel<false, false, true, false><<<dim3(DDIM / 128, (BB * NCOND) / 128), 256, 0, stream>>>(
        condb, wt, cbv + l * DDIM, vcf, nullptr, BB * NCOND, DDIM, CONDD);
    cross_attn_kernel<<<AT, 256, 0, stream>>>(qf, kcf, vcf, attf);
    f32_to_bf16_kernel<<<1024, 256, 0, stream>>>(attf, attb, BT * DDIM);
    tconv(cWo + (size_t)l * DDIM * DDIM, DDIM, DDIM);
    gemm_bf16_kernel<true, false, true, false><<<dim3(DDIM / 128, BT / 128), 256, 0, stream>>>(
        attb, wt, cbo + l * DDIM, h, nullptr, BT, DDIM, DDIM);

    // ---- FFN ----
    ln_kernel<<<BT, 256, 0, stream>>>(h, ln3_g + l * DDIM, ln3_b + l * DDIM, lnb);
    tconv(W1 + (size_t)l * DDIM * DFFN, DDIM, DFFN);
    gemm_bf16_kernel<false, true, false, true><<<dim3(DFFN / 128, BT / 128), 256, 0, stream>>>(
        lnb, wt, b1 + l * DFFN, nullptr, ffnb, BT, DFFN, DDIM);
    tconv(W2 + (size_t)l * DFFN * DDIM, DFFN, DDIM);
    gemm_bf16_kernel<true, false, true, false><<<dim3(DDIM / 128, BT / 128), 256, 0, stream>>>(
        ffnb, wt, b2 + l * DDIM, h, nullptr, BT, DDIM, DFFN);
  }

  // ---- final LN + per-codebook heads ----
  ln_kernel<<<BT, 256, 0, stream>>>(h, outg, outb, lnb);
  for (int q = 0; q < NQC; q++) {
    tconv(headW + (size_t)q * DDIM * CARDN, DDIM, CARDN);
    for (int b = 0; b < BB; b++) {
      gemm_bf16_kernel<false, false, true, false><<<dim3(CARDN / 128, TT / 128), 256, 0, stream>>>(
          lnb + (size_t)b * TT * DDIM, wt, headb + q * CARDN,
          out + ((size_t)(b * NQC + q)) * TT * CARDN, nullptr, TT, CARDN, DDIM);
    }
  }
}
